// MHA_23347442221501
// MI455X (gfx1250) — matrix operand order repaired and checked by reading, not compiled, not run on hardware
//
#include <hip/hip_runtime.h>
#include <hip/hip_bf16.h>

typedef __attribute__((ext_vector_type(16))) _Float16 v16h;
typedef __attribute__((ext_vector_type(8)))  float    v8f;

#define BN    2048
#define DD    512
#define NB    8
#define NT    8            // output-dim tiles per wave in attention (8*16 = 128 dims)
#define QSCALE 0.044194173824159216f   // 1/sqrt(512)
#define NEG_BIG (-1.0e10f)

// ---------------------------------------------------------------------------
// Kernel 1: Q = H*Wq (scaled), K^T, V projections.  One wave -> one 16x16 tile.
// z = 0:Q (row-major, scaled), 1:K (stored transposed KT[b][d][n]), 2:V (row-major)
// ---------------------------------------------------------------------------
__global__ __launch_bounds__(32) void proj_kernel(
    const float* __restrict__ H,
    const float* __restrict__ Wq, const float* __restrict__ Wk, const float* __restrict__ Wv,
    _Float16* __restrict__ Qh, _Float16* __restrict__ KTh, _Float16* __restrict__ Vh)
{
    const int lane = threadIdx.x;
    const int m0   = blockIdx.x * 16;          // row tile into flattened [B*N]
    const int n0   = blockIdx.y * 16;          // col tile into D
    const int z    = blockIdx.z;
    const float* W = (z == 0) ? Wq : (z == 1) ? Wk : Wv;

    const int nl   = lane & 15;
    const int hb   = lane >> 4;
    const int koff = hb << 3;                  // A-layout K offset per half-wave
    const float* hrow = H + (size_t)(m0 + nl) * DD;

    v8f acc = {0.f,0.f,0.f,0.f,0.f,0.f,0.f,0.f};

    for (int k0 = 0; k0 < DD; k0 += 32) {
        // A fragment: row (m0+nl), K = k0+koff+0..7 and k0+koff+16..23
        const float* ha = hrow + k0 + koff;
        float4 a0 = *(const float4*)(ha);
        float4 a1 = *(const float4*)(ha + 4);
        float4 a2 = *(const float4*)(ha + 16);
        float4 a3 = *(const float4*)(ha + 20);
        v16h A;
        A[0]=(_Float16)a0.x; A[1]=(_Float16)a0.y; A[2]=(_Float16)a0.z; A[3]=(_Float16)a0.w;
        A[4]=(_Float16)a1.x; A[5]=(_Float16)a1.y; A[6]=(_Float16)a1.z; A[7]=(_Float16)a1.w;
        A[8]=(_Float16)a2.x; A[9]=(_Float16)a2.y; A[10]=(_Float16)a2.z; A[11]=(_Float16)a2.w;
        A[12]=(_Float16)a3.x; A[13]=(_Float16)a3.y; A[14]=(_Float16)a3.z; A[15]=(_Float16)a3.w;
        // B fragment: lane holds W col n0+nl, K = k0+koff+0..7 and k0+koff+16..23
        const float* wr = W + (size_t)(k0 + koff) * DD + n0 + nl;
        v16h Bf;
        #pragma unroll
        for (int s = 0; s < 8; ++s) {
            Bf[s]     = (_Float16)wr[(size_t)s * DD];
            Bf[s + 8] = (_Float16)wr[(size_t)(s + 16) * DD];
        }

        acc = __builtin_amdgcn_wmma_f32_16x16x32_f16(false, A, false, Bf,
                                                     (short)0, acc, false, false);
    }

    // C/D layout: VGPR r -> (M = r + 8*hb, N = nl)
    const int nn    = n0 + nl;
    const int mbase = m0 + (hb << 3);
    if (z == 0) {
        #pragma unroll
        for (int r = 0; r < 8; ++r)
            Qh[(size_t)(mbase + r) * DD + nn] = (_Float16)(acc[r] * QSCALE);
    } else if (z == 2) {
        #pragma unroll
        for (int r = 0; r < 8; ++r)
            Vh[(size_t)(mbase + r) * DD + nn] = (_Float16)acc[r];
    } else {
        #pragma unroll
        for (int r = 0; r < 8; ++r) {
            const int mg = mbase + r;
            const int bb = mg >> 11;
            const int ni = mg & (BN - 1);
            KTh[((size_t)bb * DD + nn) * BN + ni] = (_Float16)acc[r];
        }
    }
}

// ---------------------------------------------------------------------------
// Kernel 2: fused flash attention. One wave per (16-query tile, batch, 128-dim
// output slice).  Key blocks of 32; online softmax; all GEMMs via WMMA f16.
// Q tile staged LDS via gfx1250 async-load path (ASYNCcnt).
// ---------------------------------------------------------------------------
__global__ __launch_bounds__(32) void attn_kernel(
    const _Float16* __restrict__ Qh, const _Float16* __restrict__ KTh,
    const _Float16* __restrict__ Vh,
    const float* __restrict__ bias, const int* __restrict__ mask,
    float* __restrict__ out)
{
    __shared__ _Float16 sQ[16 * DD];   // 16 KB query tile
    __shared__ _Float16 sP[16 * 32];   // 1 KB prob tile (C-layout -> A-layout bridge)

    const int lane = threadIdx.x;
    const int m0 = blockIdx.x * 16;
    const int b  = blockIdx.y;
    const int d0 = blockIdx.z * (NT * 16);

    // ---- Stage Q tile into LDS with async loads (memory -> LDS, no VGPR data,
    //      tracked by ASYNCcnt).  16 KB = 1024 x 16B chunks, 32 lanes.
    {
        const _Float16* src = Qh + ((size_t)b * BN + m0) * DD;
        for (int i = lane; i < 16 * DD / 8; i += 32) {
            const unsigned ldsoff = (unsigned)(size_t)(sQ) + (unsigned)i * 16u;
            const unsigned long long ga = (unsigned long long)(size_t)(src + i * 8);
            asm volatile("global_load_async_to_lds_b128 %0, %1, off"
                         :: "v"(ldsoff), "v"(ga)
                         : "memory");
        }
        asm volatile("s_wait_asynccnt 0x0" ::: "memory");
    }
    __syncthreads();

    const int nl   = lane & 15;
    const int hb   = lane >> 4;
    const int koff = hb << 3;
    const _Float16* KTb   = KTh  + (size_t)b * DD * BN;
    const _Float16* Vb    = Vh   + (size_t)b * BN * DD;
    const float*    biasb = bias + (size_t)b * BN * BN;
    const int*      maskb = mask + (size_t)b * BN * BN;

    float mrow[8], lrow[8];
    v8f accO[NT];
    const v8f vzero = {0.f,0.f,0.f,0.f,0.f,0.f,0.f,0.f};
    #pragma unroll
    for (int r = 0; r < 8; ++r) { mrow[r] = -3.0e38f; lrow[r] = 0.f; }
    #pragma unroll
    for (int t = 0; t < NT; ++t) accO[t] = vzero;

    for (int j = 0; j < BN; j += 32) {
        // ---- S = (Q*scale) @ K^T over full d : two 16x16 score tiles ----
        v8f s0 = vzero, s1 = vzero;
        for (int k0 = 0; k0 < DD; k0 += 32) {
            v16h A;
            uint4* ap = (uint4*)&A;
            const _Float16* qa = sQ + nl * DD + k0 + koff;
            ap[0] = *(const uint4*)(qa);
            ap[1] = *(const uint4*)(qa + 16);

            const _Float16* kr = KTb + (size_t)(k0 + koff) * BN + j + nl;
            v16h B0, B1;
            #pragma unroll
            for (int s = 0; s < 8; ++s) {
                B0[s]     = kr[(size_t)s * BN];
                B0[s + 8] = kr[(size_t)(s + 16) * BN];
                B1[s]     = kr[(size_t)s * BN + 16];
                B1[s + 8] = kr[(size_t)(s + 16) * BN + 16];
            }

            s0 = __builtin_amdgcn_wmma_f32_16x16x32_f16(false, A, false, B0,
                                                        (short)0, s0, false, false);
            s1 = __builtin_amdgcn_wmma_f32_16x16x32_f16(false, A, false, B1,
                                                        (short)0, s1, false, false);
        }

        // Prefetch next key-block's bias rows (gfx1250 global_prefetch_b8)
        if (j + 32 < BN)
            __builtin_prefetch(biasb + (size_t)(m0 + nl) * BN + j + 32, 0, 1);

        // ---- bias + mask + online softmax (row stats via 16-lane xor reductions)
        __syncthreads();              // protect sP reuse from previous iteration
        float alpha[8];
        #pragma unroll
        for (int r = 0; r < 8; ++r) {
            const size_t ro = (size_t)(m0 + r + (hb << 3)) * BN + j + nl;
            float v0 = s0[r] + biasb[ro];
            float v1 = s1[r] + biasb[ro + 16];
            if (maskb[ro]      == 0) v0 = NEG_BIG;
            if (maskb[ro + 16] == 0) v1 = NEG_BIG;

            float mv = fmaxf(v0, v1);
            mv = fmaxf(mv, __shfl_xor(mv, 1));
            mv = fmaxf(mv, __shfl_xor(mv, 2));
            mv = fmaxf(mv, __shfl_xor(mv, 4));
            mv = fmaxf(mv, __shfl_xor(mv, 8));
            const float mnew = fmaxf(mrow[r], mv);
            const float a  = __expf(mrow[r] - mnew);
            const float p0 = __expf(v0 - mnew);
            const float p1 = __expf(v1 - mnew);
            mrow[r] = mnew;
            float rs = p0 + p1;
            rs += __shfl_xor(rs, 1);
            rs += __shfl_xor(rs, 2);
            rs += __shfl_xor(rs, 4);
            rs += __shfl_xor(rs, 8);
            lrow[r] = lrow[r] * a + rs;
            alpha[r] = a;

            const int mr = r + (hb << 3);            // C-layout row within tile
            sP[mr * 32 + nl]      = (_Float16)p0;
            sP[mr * 32 + nl + 16] = (_Float16)p1;
        }
        __syncthreads();              // sP fully written before A-layout reads

        // ---- O = alpha*O + P @ V_block ----
        v16h PA;
        uint4* pp = (uint4*)&PA;
        const _Float16* pa = sP + nl * 32 + koff;
        pp[0] = *(const uint4*)(pa);
        pp[1] = *(const uint4*)(pa + 16);

        #pragma unroll
        for (int t = 0; t < NT; ++t) {
            #pragma unroll
            for (int r = 0; r < 8; ++r) accO[t][r] *= alpha[r];
            const _Float16* vr = Vb + (size_t)(j + koff) * DD + d0 + t * 16 + nl;
            v16h VB;
            #pragma unroll
            for (int s = 0; s < 8; ++s) {
                VB[s]     = vr[(size_t)s * DD];
                VB[s + 8] = vr[(size_t)(s + 16) * DD];
            }
            accO[t] = __builtin_amdgcn_wmma_f32_16x16x32_f16(false, PA, false, VB,
                                                             (short)0, accO[t], false, false);
        }
    }

    // ---- epilogue: normalize and store fp32 ----
    #pragma unroll
    for (int r = 0; r < 8; ++r) {
        const float inv = 1.0f / lrow[r];
        const size_t orow = ((size_t)b * BN + m0 + r + (hb << 3)) * DD + d0 + nl;
        #pragma unroll
        for (int t = 0; t < NT; ++t)
            out[orow + t * 16] = accO[t][r] * inv;
    }
}

// ---------------------------------------------------------------------------
extern "C" void kernel_launch(void* const* d_in, const int* in_sizes, int n_in,
                              void* d_out, int out_size, void* d_ws, size_t ws_size,
                              hipStream_t stream) {
    const float* H    = (const float*)d_in[0];
    const float* bias = (const float*)d_in[1];
    const int*   mask = (const int*)d_in[2];
    const float* Wq   = (const float*)d_in[3];
    const float* Wk   = (const float*)d_in[4];
    const float* Wv   = (const float*)d_in[5];
    float* out = (float*)d_out;

    _Float16* Qh  = (_Float16*)d_ws;                         // [B*N, D] f16 (pre-scaled)
    _Float16* KTh = Qh  + (size_t)NB * BN * DD;              // [B, D, N] f16 (transposed)
    _Float16* Vh  = KTh + (size_t)NB * BN * DD;              // [B*N, D] f16

    proj_kernel<<<dim3((NB * BN) / 16, DD / 16, 3), 32, 0, stream>>>(
        H, Wq, Wk, Wv, Qh, KTh, Vh);

    attn_kernel<<<dim3(BN / 16, NB, DD / (NT * 16)), 32, 0, stream>>>(
        Qh, KTh, Vh, bias, mask, out);
}
